// PolicyValueNet_37177236914529
// MI455X (gfx1250) — matrix operand order repaired and checked by reading, not compiled, not run on hardware
//
#include <hip/hip_runtime.h>
#include <hip/hip_bf16.h>
#include <math.h>

// ---------------- model dims ----------------
#define HH 20
#define WW 20
#define CCH 64
#define NPX 400
#define NMOVES 400000
#define MAXC 5
#define KDIM 96          // 83 features padded to 96 (3 WMMA k-steps of 32)
#define NOUT 128
#define TILE_M 128
#define TILES_PER_BLOCK 5
#define POLICY_BLOCKS 625   // 625*5*128 = 400000

typedef _Float16 v16h __attribute__((ext_vector_type(16)));
typedef _Float16 h8   __attribute__((ext_vector_type(8)));
typedef float    v8f  __attribute__((ext_vector_type(8)));

union H16 { h8 h[2]; _Float16 e[16]; v16h v; };

// ---------------- encoder: generic 3x3 conv + folded BN + (residual) + relu ----
__global__ __launch_bounds__(256)
void conv3x3_bn_relu(const float* __restrict__ in, const float* __restrict__ wgt,
                     const float* __restrict__ bias, const float* __restrict__ scale,
                     const float* __restrict__ off, const float* __restrict__ res,
                     float* __restrict__ out, int cin) {
  int idx = blockIdx.x * blockDim.x + threadIdx.x;
  if (idx >= CCH * NPX) return;
  int co = idx / NPX;
  int p  = idx % NPX;
  int y = p / WW, x = p % WW;
  const float* wb = wgt + co * cin * 9;
  float acc = 0.f;
  for (int ci = 0; ci < cin; ++ci) {
    const float* ip = in + ci * NPX;
    const float* wp = wb + ci * 9;
    #pragma unroll
    for (int dy = -1; dy <= 1; ++dy) {
      int yy = y + dy;
      if (yy < 0 || yy >= HH) continue;
      #pragma unroll
      for (int dx = -1; dx <= 1; ++dx) {
        int xx = x + dx;
        if (xx < 0 || xx >= WW) continue;
        acc += ip[yy * WW + xx] * wp[(dy + 1) * 3 + (dx + 1)];
      }
    }
  }
  float v = (acc + bias[co]) * scale[co] + off[co];
  if (res) v += res[idx];
  out[idx] = fmaxf(v, 0.f);
}

// ---------------- pack fmap [c][y][x] f32 -> [y][x][c] f16 -------------------
__global__ __launch_bounds__(256)
void pack_fmap_f16(const float* __restrict__ fmap, _Float16* __restrict__ out) {
  int idx = blockIdx.x * blockDim.x + threadIdx.x;
  if (idx >= CCH * NPX) return;
  int c = idx & 63;
  int p = idx >> 6;
  out[p * CCH + c] = (_Float16)fmap[c * NPX + p];
}

// ---------------- value head (one tiny block) --------------------------------
__global__ __launch_bounds__(256)
void value_head(const float* __restrict__ fmap,        // [64][400] f32
                const float* __restrict__ vw, const float* __restrict__ vb,
                const float* __restrict__ self_rem, const float* __restrict__ opp_rem,
                const float* __restrict__ w1, const float* __restrict__ b1,
                const float* __restrict__ w2, const float* __restrict__ b2,
                float* __restrict__ out_value) {
  __shared__ float red[256];
  __shared__ float vin[74];
  __shared__ float hbuf[64];
  int tid = threadIdx.x;
  int oc = tid & 31;        // 0..31 output channels of 1x1 conv
  int sl = tid >> 5;        // 8 pixel slices of 50
  float s = 0.f;
  for (int px = sl * 50; px < sl * 50 + 50; ++px) {
    float v = vb[oc];
    for (int c = 0; c < CCH; ++c) v += fmap[c * NPX + px] * vw[oc * CCH + c];
    s += fmaxf(v, 0.f);
  }
  red[tid] = s;
  __syncthreads();
  if (tid < 32) {
    float p = 0.f;
    #pragma unroll
    for (int q = 0; q < 8; ++q) p += red[q * 32 + tid];
    vin[tid] = p * (1.0f / 400.0f);
  }
  if (tid >= 32 && tid < 53) vin[tid] = self_rem[tid - 32];
  if (tid >= 53 && tid < 74) vin[tid] = opp_rem[tid - 53];
  __syncthreads();
  if (tid < 64) {
    float hv = b1[tid];
    for (int j = 0; j < 74; ++j) hv += vin[j] * w1[j * 64 + tid];
    hbuf[tid] = fmaxf(hv, 0.f);
  }
  __syncthreads();
  if (tid == 0) {
    float v = b2[0];
    for (int k = 0; k < 64; ++k) v += hbuf[k] * w2[k];
    out_value[0] = tanhf(v);
  }
}

// ---------------- policy head: gather-mean + WMMA GEMM + fused epilogue ------
// Transposed GEMM: D^T[128 hidden, 128 moves] = w1^T x features^T.
// Wave wv holds A fragments (w1 rows 16wv..16wv+15) in registers and loops
// over 8 move-tiles. Output layout: lane holds one move column, its 8
// accumulator VGPRs hold 8 hidden units -> epilogue reduction is in-lane.
__global__ __launch_bounds__(256)
void policy_head(const _Float16* __restrict__ fmap_h,   // [y][x][c] f16
                 const int* __restrict__ piece_id,
                 const float* __restrict__ anchor,
                 const float* __restrict__ sizef,
                 const int* __restrict__ num_cells,
                 const int* __restrict__ cells,
                 const float* __restrict__ piece_table,
                 const float* __restrict__ w1,          // [83][128] f32
                 const float* __restrict__ b1,          // [128]
                 const float* __restrict__ w2,          // [128]
                 const float* __restrict__ b2,          // [1]
                 float* __restrict__ logits) {
  // double-buffered feature tiles so tile t+1's gather overlaps tile t's WMMAs
  __shared__ _Float16 sh_feat[2][TILE_M * KDIM];  // 2 x 24 KB
  __shared__ float    sh_part[8 * TILE_M];        // per-wave partial logits, 4 KB

  const int tid  = threadIdx.x;
  const int lane = tid & 31;
  const int wv   = tid >> 5;          // 8 waves; wave wv owns hidden rows [16wv,16wv+16)
  const int g    = lane >> 4;         // lane group
  const int r15  = lane & 15;

  const int   ln  = wv * 16 + r15;    // this lane's hidden row (A-matrix M index)
  const float b2v = b2[0];

  // A fragments (w1 rows for this wave) in registers, loop-invariant.
  // 16-bit A 16x32 layout: lane m=l%16; halves 0..7 -> K=8g+h, 8..15 -> K=16+8g+(h-8).
  H16 Af[3];
  #pragma unroll
  for (int kb = 0; kb < 3; ++kb) {
    #pragma unroll
    for (int h = 0; h < 16; ++h) {
      int K = kb * 32 + (h < 8 ? 8 * g + h : 16 + 8 * g + (h - 8));
      float v = (K < 83) ? w1[K * NOUT + ln] : 0.f;
      Af[kb].e[h] = (_Float16)v;
    }
  }
  // per-lane bias / w2 for the 8 hidden units this lane's accumulators hold:
  // D row (M) = r + 8g within the wave's 16-row tile.
  float b1r[8], w2r[8];
  #pragma unroll
  for (int r = 0; r < 8; ++r) {
    int n = wv * 16 + g * 8 + r;
    b1r[r] = b1[n];
    w2r[r] = w2[n];
  }

  // gather/feature-build for one tile into sh_feat[buf] (2 threads per move,
  // packed f16 accumulation -> v_pk_add_f16)
  auto gather_tile = [&](int move_base, int buf) {
    const int slot = tid >> 1;      // 0..127 local move
    const int hs   = tid & 1;       // channel half: 0 -> c[0..31], 1 -> c[32..63]
    const int gm   = move_base + slot;
    const int nc   = num_cells[gm];
    h8 acc4[4] = {};
    #pragma unroll
    for (int j = 0; j < MAXC; ++j) {
      if (j < nc) {
        int2 cxy = *(const int2*)(cells + (gm * MAXC + j) * 2);  // (x, y)
        const _Float16* cp = fmap_h + (cxy.y * WW + cxy.x) * CCH + hs * 32;
        #pragma unroll
        for (int q = 0; q < 4; ++q) acc4[q] += *(const h8*)(cp + q * 8);
      }
    }
    const _Float16 invh = (_Float16)(1.0f / (float)nc);
    h8 invv;
    #pragma unroll
    for (int e = 0; e < 8; ++e) invv[e] = invh;
    _Float16* fr = sh_feat[buf] + slot * KDIM;
    #pragma unroll
    for (int q = 0; q < 4; ++q) *(h8*)(fr + hs * 32 + q * 8) = acc4[q] * invv;
    if (hs == 0) {
      const float* pt = piece_table + piece_id[gm] * 16;
      #pragma unroll
      for (int e = 0; e < 16; ++e) fr[64 + e] = (_Float16)pt[e];
      fr[80] = (_Float16)anchor[gm * 2 + 0];
      fr[81] = (_Float16)anchor[gm * 2 + 1];
      fr[82] = (_Float16)sizef[gm];
    } else {
      #pragma unroll
      for (int e = 83; e < KDIM; ++e) fr[e] = (_Float16)0.f;
    }
  };

  const int tile0_base = blockIdx.x * TILES_PER_BLOCK * TILE_M;
  gather_tile(tile0_base, 0);
  __syncthreads();

  for (int t = 0; t < TILES_PER_BLOCK; ++t) {
    const int move_base = tile0_base + t * TILE_M;

    // ---- combined region: prefetch-gather tile t+1 + WMMAs for tile t ----
    if (t + 1 < TILES_PER_BLOCK) gather_tile(move_base + TILE_M, (t + 1) & 1);

    {
      v8f acc[8] = {};
      const _Float16* featb = sh_feat[t & 1];
      #pragma unroll
      for (int kb = 0; kb < 3; ++kb) {
        #pragma unroll
        for (int mt = 0; mt < 8; ++mt) {
          // B fragment: feature rows as columns. Column = move mt*16 + r15;
          // halves 0..7 -> K = kb*32 + 8g + h, 8..15 -> K = kb*32 + 16 + 8g + (h-8).
          H16 B;
          const _Float16* rp = featb + (mt * 16 + r15) * KDIM + kb * 32 + g * 8;
          B.h[0] = *(const h8*)(rp);
          B.h[1] = *(const h8*)(rp + 16);
          acc[mt] = __builtin_amdgcn_wmma_f32_16x16x32_f16(
              false, Af[kb].v, false, B.v, (short)0, acc[mt], false, false);
        }
      }
      // fused epilogue: in-lane relu(h + b1) * w2 over this lane's 8 hidden
      // units, one cross-group shuffle, per-wave partial into LDS.
      #pragma unroll
      for (int mt = 0; mt < 8; ++mt) {
        float p = 0.f;
        #pragma unroll
        for (int r = 0; r < 8; ++r) {
          float hv = acc[mt][r] + b1r[r];
          hv = hv > 0.f ? hv : 0.f;
          p = fmaf(hv, w2r[r], p);
        }
        p += __shfl_xor(p, 16, 32);   // combine the two 8-row lane groups
        if (g == 0) sh_part[wv * TILE_M + mt * 16 + r15] = p;
      }
    }
    __syncthreads();

    // ---- cross-wave sum + bias, write logits ----
    if (tid < TILE_M) {
      float s = b2v;
      #pragma unroll
      for (int w8 = 0; w8 < 8; ++w8) s += sh_part[w8 * TILE_M + tid];
      logits[move_base + tid] = s;
    }
    __syncthreads();
  }
}

// ---------------- launcher ----------------------------------------------------
extern "C" void kernel_launch(void* const* d_in, const int* in_sizes, int n_in,
                              void* d_out, int out_size, void* d_ws, size_t ws_size,
                              hipStream_t stream) {
  (void)in_sizes; (void)n_in; (void)out_size; (void)ws_size;

  const float* board      = (const float*)d_in[0];
  const float* self_rem   = (const float*)d_in[1];
  const float* opp_rem    = (const float*)d_in[2];
  const int*   piece_id   = (const int*)  d_in[3];
  const float* anchor     = (const float*)d_in[4];
  const int*   num_cells  = (const int*)  d_in[5];
  const float* sizef      = (const float*)d_in[6];
  const int*   cells      = (const int*)  d_in[7];
  const float* stem_w     = (const float*)d_in[8];
  const float* stem_b     = (const float*)d_in[9];
  const float* stem_s     = (const float*)d_in[10];
  const float* stem_o     = (const float*)d_in[11];
  const float* blk_w1     = (const float*)d_in[12];
  const float* blk_b1     = (const float*)d_in[13];
  const float* blk_s1     = (const float*)d_in[14];
  const float* blk_o1     = (const float*)d_in[15];
  const float* blk_w2     = (const float*)d_in[16];
  const float* blk_b2     = (const float*)d_in[17];
  const float* blk_s2     = (const float*)d_in[18];
  const float* blk_o2     = (const float*)d_in[19];
  const float* piece_tab  = (const float*)d_in[20];
  const float* mlp_w1     = (const float*)d_in[21];
  const float* mlp_b1     = (const float*)d_in[22];
  const float* mlp_w2     = (const float*)d_in[23];
  const float* mlp_b2     = (const float*)d_in[24];
  const float* val_conv_w = (const float*)d_in[25];
  const float* val_conv_b = (const float*)d_in[26];
  const float* val_w1     = (const float*)d_in[27];
  const float* val_b1     = (const float*)d_in[28];
  const float* val_w2     = (const float*)d_in[29];
  const float* val_b2     = (const float*)d_in[30];

  float* out = (float*)d_out;             // [0..400000) logits, [400000] value

  // workspace: two f32 ping-pong fmaps + packed f16 fmap
  float*     bufA   = (float*)d_ws;       // 25600 f32
  float*     bufB   = bufA + CCH * NPX;   // 25600 f32
  _Float16*  fmap_h = (_Float16*)(bufB + CCH * NPX); // 25600 f16

  dim3 cgrid(100), cblk(256);
  // stem
  conv3x3_bn_relu<<<cgrid, cblk, 0, stream>>>(board, stem_w, stem_b, stem_s, stem_o,
                                              nullptr, bufA, 5);
  // 4 residual blocks
  for (int i = 0; i < 4; ++i) {
    conv3x3_bn_relu<<<cgrid, cblk, 0, stream>>>(bufA, blk_w1 + i * CCH * CCH * 9,
                                                blk_b1 + i * CCH, blk_s1 + i * CCH,
                                                blk_o1 + i * CCH, nullptr, bufB, CCH);
    conv3x3_bn_relu<<<cgrid, cblk, 0, stream>>>(bufB, blk_w2 + i * CCH * CCH * 9,
                                                blk_b2 + i * CCH, blk_s2 + i * CCH,
                                                blk_o2 + i * CCH, bufA, bufA, CCH);
  }
  pack_fmap_f16<<<cgrid, cblk, 0, stream>>>(bufA, fmap_h);

  value_head<<<1, 256, 0, stream>>>(bufA, val_conv_w, val_conv_b, self_rem, opp_rem,
                                    val_w1, val_b1, val_w2, val_b2, out + NMOVES);

  policy_head<<<POLICY_BLOCKS, 256, 0, stream>>>(fmap_h, piece_id, anchor, sizef,
                                                 num_cells, cells, piece_tab,
                                                 mlp_w1, mlp_b1, mlp_w2, mlp_b2, out);
}
